// RWKV_X_52956946760514
// MI455X (gfx1250) — matrix operand order repaired and checked by reading, not compiled, not run on hardware
//
#include <hip/hip_runtime.h>

typedef __attribute__((ext_vector_type(16))) _Float16 v16h;
typedef __attribute__((ext_vector_type(8)))  _Float16 v8h;
typedef __attribute__((ext_vector_type(4)))  _Float16 v4h;
typedef __attribute__((ext_vector_type(8)))  float    v8f;
typedef __attribute__((ext_vector_type(4)))  float    v4f;

#define T_LEN 1024
#define C_DIM 1024
#define H_HEADS 16
#define HN 64

__device__ __forceinline__ float sigmoidf_(float x) { return 1.0f / (1.0f + __expf(-x)); }

// ---------------------------------------------------------------------------
// Generic WMMA GEMM: D[M,N] = epilogue(A[M,K] @ B[K,N])
// fp32 in/out, fp16 WMMA compute (v_wmma_f32_16x16x32_f16), fp32 accumulate.
// Block tile 128(M) x 32(N), K-step 32, double-buffered LDS (one barrier/step).
// 8 waves; wave w owns M-subtile w and BOTH 16-wide N-subtiles (2 accumulators,
// A fragment reused across 2 WMMAs; B tile reused by all 8 waves).
// Staging: A via float4 (global_load_b128) -> half4 (ds_store_b64);
//          B via 4 coalesced b32 -> transposed half4 (ds_store_b64).
// MODE: 0=store 1=tanh 2=sigmoid 3=sigmoid(bias[col]+v) 4=relu^2 5=extra[idx]+v
// ---------------------------------------------------------------------------
template <int MODE>
__global__ __launch_bounds__(256) void wmma_gemm_kernel(
    const float* __restrict__ A, const float* __restrict__ B,
    float* __restrict__ D, const float* __restrict__ extra,
    int M, int N, int K)
{
  // Row stride 48 halves (96 B): multiple of 16 B so v8h LDS reads stay aligned.
  __shared__ __align__(32) _Float16 As[2][128][48];
  __shared__ __align__(32) _Float16 Bt[2][32][48];   // transposed: Bt[n][k]

  const int tid  = threadIdx.x;
  const int lane = tid & 31;
  const int wave = tid >> 5;           // 0..7 -> M sub-tile
  const int m0   = blockIdx.y * 128;
  const int n0   = blockIdx.x * 32;
  const int h    = lane >> 4;          // K-half select (contiguous: k = 16h + s)
  const int l15  = lane & 15;

  // B staging coords: lane-consecutive n -> coalesced 128B row segments.
  const int bn = tid & 31;
  const int bk = (tid >> 5) * 4;

  auto stage = [&](int k0, int buf, bool pref) {
    // A tile: 128x32 fp32; 4 float4 per thread.
    v4f av[4];
    #pragma unroll
    for (int it = 0; it < 4; ++it) {
      int f4 = tid + it * 256;           // 1024 float4s
      int r  = f4 >> 3;                  // 8 float4 per 32-float row
      int c  = (f4 & 7) * 4;
      av[it] = *(const v4f*)&A[(size_t)(m0 + r) * K + (k0 + c)];
    }
    // B tile: 32x32 fp32; 4 coalesced b32 per thread (same n, k..k+3).
    float b0 = B[(size_t)(k0 + bk + 0) * N + (n0 + bn)];
    float b1 = B[(size_t)(k0 + bk + 1) * N + (n0 + bn)];
    float b2 = B[(size_t)(k0 + bk + 2) * N + (n0 + bn)];
    float b3 = B[(size_t)(k0 + bk + 3) * N + (n0 + bn)];
    if (pref) {  // wave-uniform guard: speculative prefetch of K+1 tile
      #pragma unroll
      for (int it = 0; it < 4; ++it) {
        int f4 = tid + it * 256;
        int r  = f4 >> 3;
        int c  = (f4 & 7) * 4;
        __builtin_prefetch(&A[(size_t)(m0 + r) * K + (k0 + 32 + c)], 0, 1);
      }
    }
    #pragma unroll
    for (int it = 0; it < 4; ++it) {
      int f4 = tid + it * 256;
      int r  = f4 >> 3;
      int c  = (f4 & 7) * 4;
      v4h ah;
      ah[0] = (_Float16)av[it][0]; ah[1] = (_Float16)av[it][1];
      ah[2] = (_Float16)av[it][2]; ah[3] = (_Float16)av[it][3];
      *(v4h*)&As[buf][r][c] = ah;
    }
    v4h bh;
    bh[0] = (_Float16)b0; bh[1] = (_Float16)b1;
    bh[2] = (_Float16)b2; bh[3] = (_Float16)b3;
    *(v4h*)&Bt[buf][bn][bk] = bh;        // transposed scatter, single b64 store
  };

  v8f acc0 = {};
  v8f acc1 = {};
  const int KT = K >> 5;

  stage(0, 0, KT > 1);
  __syncthreads();

  for (int kt = 0; kt < KT; ++kt) {
    const int cur = kt & 1;
    if (kt + 1 < KT) stage((kt + 1) << 5, cur ^ 1, kt + 2 < KT);

    // A fragment: lane half h covers K = h*16..h*16+15 (contiguous, matches B)
    v8h a_lo = *(const v8h*)&As[cur][wave * 16 + l15][h * 16];
    v8h a_hi = *(const v8h*)&As[cur][wave * 16 + l15][h * 16 + 8];
    // B fragments: lane half h covers K = h*16..h*16+15 (contiguous in Bt row)
    v8h b0lo = *(const v8h*)&Bt[cur][l15][h * 16];
    v8h b0hi = *(const v8h*)&Bt[cur][l15][h * 16 + 8];
    v8h b1lo = *(const v8h*)&Bt[cur][16 + l15][h * 16];
    v8h b1hi = *(const v8h*)&Bt[cur][16 + l15][h * 16 + 8];
    v16h af, bf0, bf1;
    #pragma unroll
    for (int e = 0; e < 8; ++e) {
      af[e]  = a_lo[e]; af[e + 8]  = a_hi[e];
      bf0[e] = b0lo[e]; bf0[e + 8] = b0hi[e];
      bf1[e] = b1lo[e]; bf1[e + 8] = b1hi[e];
    }
    acc0 = __builtin_amdgcn_wmma_f32_16x16x32_f16(false, af, false, bf0,
                                                  (short)0, acc0, false, false);
    acc1 = __builtin_amdgcn_wmma_f32_16x16x32_f16(false, af, false, bf1,
                                                  (short)0, acc1, false, false);
    __syncthreads();
  }

  // Epilogue. C/D layout: lane l15 = column, VGPR r -> row (h*8 + r).
  const int rbase = m0 + wave * 16 + h * 8;
  const int col0  = n0 + l15;
  const int col1  = n0 + 16 + l15;
  #pragma unroll
  for (int r = 0; r < 8; ++r) {
    int row = rbase + r;
    float v0_ = acc0[r];
    float v1_ = acc1[r];
    if (MODE == 1)      { v0_ = tanhf(v0_); v1_ = tanhf(v1_); }
    else if (MODE == 2) { v0_ = sigmoidf_(v0_); v1_ = sigmoidf_(v1_); }
    else if (MODE == 3) { v0_ = sigmoidf_(extra[col0] + v0_);
                          v1_ = sigmoidf_(extra[col1] + v1_); }
    else if (MODE == 4) { float t0 = fmaxf(v0_, 0.f); v0_ = t0 * t0;
                          float t1 = fmaxf(v1_, 0.f); v1_ = t1 * t1; }
    else if (MODE == 5) { v0_ = extra[(size_t)row * N + col0] + v0_;
                          v1_ = extra[(size_t)row * N + col1] + v1_; }
    D[(size_t)row * N + col0] = v0_;
    D[(size_t)row * N + col1] = v1_;
  }
}

// ---------------------------------------------------------------------------
// LayerNorm over C=1024 per token (one block per token).
// ---------------------------------------------------------------------------
__global__ __launch_bounds__(256) void ln_kernel(
    const float* __restrict__ x, const float* __restrict__ w,
    const float* __restrict__ b, float* __restrict__ out)
{
  __shared__ float ssum[256];
  __shared__ float ssq[256];
  const int t = blockIdx.x;
  const float* xp = x + (size_t)t * C_DIM;
  float s = 0.f, q = 0.f;
  for (int c = threadIdx.x; c < C_DIM; c += 256) { float v = xp[c]; s += v; q += v * v; }
  ssum[threadIdx.x] = s; ssq[threadIdx.x] = q;
  __syncthreads();
  for (int off = 128; off > 0; off >>= 1) {
    if (threadIdx.x < off) {
      ssum[threadIdx.x] += ssum[threadIdx.x + off];
      ssq[threadIdx.x]  += ssq[threadIdx.x + off];
    }
    __syncthreads();
  }
  float m   = ssum[0] * (1.f / C_DIM);
  float var = ssq[0] * (1.f / C_DIM) - m * m;
  float rs  = rsqrtf(var + 1e-5f);
  float* op = out + (size_t)t * C_DIM;
  for (int c = threadIdx.x; c < C_DIM; c += 256) op[c] = (xp[c] - m) * rs * w[c] + b[c];
}

// ---------------------------------------------------------------------------
// Token-shift mixes.
// ---------------------------------------------------------------------------
__global__ void mix6_kernel(const float* __restrict__ xn, const float* __restrict__ prev0,
    const float* __restrict__ mr, const float* __restrict__ mw, const float* __restrict__ mk,
    const float* __restrict__ mv, const float* __restrict__ ma, const float* __restrict__ mg,
    float* __restrict__ xr, float* __restrict__ xw, float* __restrict__ xk,
    float* __restrict__ xv, float* __restrict__ xa, float* __restrict__ xg)
{
  int idx = blockIdx.x * blockDim.x + threadIdx.x;
  int t = idx >> 10, c = idx & (C_DIM - 1);
  float v = xn[idx];
  float p = t ? xn[idx - C_DIM] : prev0[c];
  float d = p - v;
  xr[idx] = v + d * mr[c]; xw[idx] = v + d * mw[c]; xk[idx] = v + d * mk[c];
  xv[idx] = v + d * mv[c]; xa[idx] = v + d * ma[c]; xg[idx] = v + d * mg[c];
}

__global__ void mix1_kernel(const float* __restrict__ xn, const float* __restrict__ prev0,
                            const float* __restrict__ mk, float* __restrict__ out)
{
  int idx = blockIdx.x * blockDim.x + threadIdx.x;
  int t = idx >> 10, c = idx & (C_DIM - 1);
  float v = xn[idx];
  float p = t ? xn[idx - C_DIM] : prev0[c];
  out[idx] = v + (p - v) * mk[c];
}

// ---------------------------------------------------------------------------
// Post-GEMM elementwise: kk normalization (per head), k/v/w adjustments.
// grid = (H, T), block = 64 (one lane per channel-in-head).
// ---------------------------------------------------------------------------
__global__ __launch_bounds__(64) void post_kernel(
    const float* __restrict__ kraw, const float* __restrict__ abuf,
    const float* __restrict__ vraw, const float* __restrict__ sv,
    const float* __restrict__ wlora, const float* __restrict__ vfirst,
    const float* __restrict__ k_k, const float* __restrict__ k_a,
    const float* __restrict__ w0,
    float* __restrict__ kk, float* __restrict__ kb,
    float* __restrict__ vb, float* __restrict__ wb)
{
  const int hh = blockIdx.x, t = blockIdx.y, n = threadIdx.x;
  const int c = hh * HN + n;
  const size_t idx = (size_t)t * C_DIM + c;
  float kr  = kraw[idx];
  float kkv = kr * k_k[c];
  __shared__ float sh[HN];
  sh[n] = kkv * kkv;
  __syncthreads();
  float ss = 0.f;
  #pragma unroll
  for (int j = 0; j < HN; ++j) ss += sh[j];
  float nrm = fmaxf(sqrtf(ss), 1e-12f);
  kk[idx] = kkv / nrm;
  float a = abuf[idx];
  kb[idx] = kr * (1.f + (a - 1.f) * k_a[c]);
  float vr = vraw[idx];
  vb[idx] = vr + (vfirst[idx] - vr) * sv[idx];
  wb[idx] = __expf(-0.606531f * sigmoidf_(w0[c] + wlora[idx]));
}

// ---------------------------------------------------------------------------
// Sequential delta-rule scan. One block per head, 64 threads; thread i keeps
// state row i (64 fp32) entirely in VGPRs. Rank-1 update per step:
//   st = st*w - (st.kk) (x) (kk*a) + v (x) k ;  y_i = st_i . r
// ---------------------------------------------------------------------------
__global__ __launch_bounds__(64) void scan_kernel(
    const float* __restrict__ wb, const float* __restrict__ kkb,
    const float* __restrict__ ab, const float* __restrict__ vb,
    const float* __restrict__ kb, const float* __restrict__ rb,
    const float* __restrict__ state0,
    float* __restrict__ yraw, float* __restrict__ state_out)
{
  const int hh = blockIdx.x;
  const int i  = threadIdx.x;
  float st[HN];
  #pragma unroll
  for (int j = 0; j < HN; ++j)
    st[j] = state0[(size_t)hh * HN * HN + (size_t)i * HN + j];

  __shared__ float s_kk[HN], s_ka[HN], s_w[HN], s_k[HN], s_r[HN];
  for (int t = 0; t < T_LEN; ++t) {
    const size_t base = (size_t)t * C_DIM + hh * HN;
    float kk_i = kkb[base + i];
    s_kk[i] = kk_i;
    s_ka[i] = kk_i * ab[base + i];
    s_w[i]  = wb[base + i];
    s_k[i]  = kb[base + i];
    s_r[i]  = rb[base + i];
    float v_i = vb[base + i];
    __syncthreads();
    float u = 0.f;
    #pragma unroll
    for (int j = 0; j < HN; ++j) u += st[j] * s_kk[j];
    float y = 0.f;
    #pragma unroll
    for (int j = 0; j < HN; ++j) {
      float sj = st[j] * s_w[j] - u * s_ka[j] + v_i * s_k[j];
      st[j] = sj;
      y += sj * s_r[j];
    }
    yraw[base + i] = y;
    __syncthreads();
  }
  #pragma unroll
  for (int j = 0; j < HN; ++j)
    state_out[(size_t)hh * HN * HN + (size_t)i * HN + j] = st[j];
}

// ---------------------------------------------------------------------------
// GroupNorm (per token, per head, eps=0.00064) + r.k bonus + gate multiply.
// ---------------------------------------------------------------------------
__global__ __launch_bounds__(64) void gnorm_kernel(
    const float* __restrict__ yraw, const float* __restrict__ rbuf,
    const float* __restrict__ kb, const float* __restrict__ vb,
    const float* __restrict__ gbuf, const float* __restrict__ lnxw,
    const float* __restrict__ lnxb, const float* __restrict__ r_k,
    float* __restrict__ ybuf)
{
  const int hh = blockIdx.x, t = blockIdx.y, n = threadIdx.x;
  const int c = hh * HN + n;
  const size_t idx = (size_t)t * C_DIM + c;
  __shared__ float sv_[HN], sb_[HN];
  float yv = yraw[idx];
  sv_[n] = yv;
  sb_[n] = rbuf[idx] * kb[idx] * r_k[c];
  __syncthreads();
  float sum = 0.f, sq = 0.f, bon = 0.f;
  #pragma unroll
  for (int j = 0; j < HN; ++j) { float v = sv_[j]; sum += v; sq += v * v; bon += sb_[j]; }
  float m   = sum * (1.f / HN);
  float var = sq * (1.f / HN) - m * m;
  float yn  = (yv - m) * rsqrtf(var + 0.00064f) * lnxw[c] + lnxb[c];
  yn += bon * vb[idx];
  ybuf[idx] = yn * gbuf[idx];
}

// ---------------------------------------------------------------------------
extern "C" void kernel_launch(void* const* d_in, const int* in_sizes, int n_in,
                              void* d_out, int out_size, void* d_ws, size_t ws_size,
                              hipStream_t stream)
{
  (void)in_sizes; (void)n_in; (void)out_size; (void)ws_size;
  const float* x       = (const float*)d_in[0];
  const float* xprev_a = (const float*)d_in[1];
  const float* xprev_f = (const float*)d_in[2];
  const float* state0  = (const float*)d_in[3];
  const float* v_first = (const float*)d_in[4];
  const float* ln1_w = (const float*)d_in[5];
  const float* ln1_b = (const float*)d_in[6];
  const float* ln2_w = (const float*)d_in[7];
  const float* ln2_b = (const float*)d_in[8];
  const float* x_r = (const float*)d_in[9];
  const float* x_w = (const float*)d_in[10];
  const float* x_k = (const float*)d_in[11];
  const float* x_v = (const float*)d_in[12];
  const float* x_a = (const float*)d_in[13];
  const float* x_g = (const float*)d_in[14];
  const float* w0 = (const float*)d_in[15];
  const float* w1 = (const float*)d_in[16];
  const float* w2 = (const float*)d_in[17];
  const float* a0 = (const float*)d_in[18];
  const float* a1 = (const float*)d_in[19];
  const float* a2 = (const float*)d_in[20];
  const float* v0 = (const float*)d_in[21];
  const float* v1 = (const float*)d_in[22];
  const float* v2 = (const float*)d_in[23];
  const float* g1 = (const float*)d_in[24];
  const float* g2 = (const float*)d_in[25];
  const float* k_k = (const float*)d_in[26];
  const float* k_a = (const float*)d_in[27];
  const float* r_k = (const float*)d_in[28];
  const float* R_ = (const float*)d_in[29];
  const float* K_ = (const float*)d_in[30];
  const float* V_ = (const float*)d_in[31];
  const float* O_ = (const float*)d_in[32];
  const float* lnx_w = (const float*)d_in[33];
  const float* lnx_b = (const float*)d_in[34];
  const float* ffn_xk = (const float*)d_in[35];
  const float* ffnK = (const float*)d_in[36];
  const float* ffnV = (const float*)d_in[37];

  const size_t TC = (size_t)T_LEN * C_DIM;
  float* out_x     = (float*)d_out;
  float* out_state = out_x + TC;

  float* ws = (float*)d_ws;
  auto S = [&](int i) { return ws + (size_t)i * TC; };
  float* xn1  = S(0);
  float* xr_  = S(1);  float* yraw = S(1);   // reuse after r GEMM
  float* xw_  = S(2);  float* ybuf = S(2);   // reuse after w LoRA
  float* xk_  = S(3);  float* xcur = S(3);   // reuse after k GEMM
  float* xv_  = S(4);  float* xn2  = S(4);   // reuse after v GEMMs
  float* xa_  = S(5);  float* kf   = S(5);   // reuse after a LoRA
  float* xg_  = S(6);  float* kkb  = S(6);   // reuse after g LoRA
  float* rbuf  = S(7);
  float* kraw  = S(8);
  float* vraw  = S(9);
  float* wlora = S(10);
  float* abuf  = S(11);
  float* svb   = S(12);
  float* gbuf  = S(13);
  float* kb    = S(14);
  float* vb    = S(15);
  float* wb    = S(16);
  float* sm    = S(17);                       // small LoRA intermediates
  float* t_w = sm;
  float* t_a = sm + (size_t)T_LEN * 64;
  float* t_v = sm + (size_t)T_LEN * 128;
  float* t_g = sm + (size_t)T_LEN * 160;
  float* hbuf = S(18);                        // T x 4096 FFN hidden (4 slots)

  const dim3 blk(256);
  auto gg = [](int M, int N) { return dim3((unsigned)(N / 32), (unsigned)(M / 128)); };

  // ---- TMix ----
  ln_kernel<<<T_LEN, 256, 0, stream>>>(x, ln1_w, ln1_b, xn1);
  mix6_kernel<<<(T_LEN * C_DIM) / 256, 256, 0, stream>>>(
      xn1, xprev_a, x_r, x_w, x_k, x_v, x_a, x_g, xr_, xw_, xk_, xv_, xa_, xg_);

  wmma_gemm_kernel<0><<<gg(1024, 1024), blk, 0, stream>>>(xr_, R_, rbuf, nullptr, 1024, 1024, 1024);
  wmma_gemm_kernel<1><<<gg(1024,   64), blk, 0, stream>>>(xw_, w1, t_w,  nullptr, 1024,   64, 1024);
  wmma_gemm_kernel<0><<<gg(1024, 1024), blk, 0, stream>>>(t_w, w2, wlora, nullptr, 1024, 1024,   64);
  wmma_gemm_kernel<0><<<gg(1024, 1024), blk, 0, stream>>>(xk_, K_, kraw, nullptr, 1024, 1024, 1024);
  wmma_gemm_kernel<0><<<gg(1024,   64), blk, 0, stream>>>(xa_, a1, t_a,  nullptr, 1024,   64, 1024);
  wmma_gemm_kernel<3><<<gg(1024, 1024), blk, 0, stream>>>(t_a, a2, abuf, a0,      1024, 1024,   64);
  wmma_gemm_kernel<0><<<gg(1024, 1024), blk, 0, stream>>>(xv_, V_, vraw, nullptr, 1024, 1024, 1024);
  wmma_gemm_kernel<0><<<gg(1024,   32), blk, 0, stream>>>(xv_, v1, t_v,  nullptr, 1024,   32, 1024);
  wmma_gemm_kernel<3><<<gg(1024, 1024), blk, 0, stream>>>(t_v, v2, svb,  v0,      1024, 1024,   32);
  wmma_gemm_kernel<2><<<gg(1024,  128), blk, 0, stream>>>(xg_, g1, t_g,  nullptr, 1024,  128, 1024);
  wmma_gemm_kernel<0><<<gg(1024, 1024), blk, 0, stream>>>(t_g, g2, gbuf, nullptr, 1024, 1024,  128);

  post_kernel<<<dim3(H_HEADS, T_LEN), 64, 0, stream>>>(
      kraw, abuf, vraw, svb, wlora, v_first, k_k, k_a, w0, kkb, kb, vb, wb);

  scan_kernel<<<H_HEADS, 64, 0, stream>>>(wb, kkb, abuf, vb, kb, rbuf, state0, yraw, out_state);

  gnorm_kernel<<<dim3(H_HEADS, T_LEN), 64, 0, stream>>>(
      yraw, rbuf, kb, vb, gbuf, lnx_w, lnx_b, r_k, ybuf);

  // att projection fused with residual add: xcur = x + (y*g) @ O_
  wmma_gemm_kernel<5><<<gg(1024, 1024), blk, 0, stream>>>(ybuf, O_, xcur, x, 1024, 1024, 1024);

  // ---- CMix ----
  ln_kernel<<<T_LEN, 256, 0, stream>>>(xcur, ln2_w, ln2_b, xn2);
  mix1_kernel<<<(T_LEN * C_DIM) / 256, 256, 0, stream>>>(xn2, xprev_f, ffn_xk, kf);
  wmma_gemm_kernel<4><<<gg(1024, 4096), blk, 0, stream>>>(kf, ffnK, hbuf, nullptr, 1024, 4096, 1024);
  wmma_gemm_kernel<5><<<gg(1024, 1024), blk, 0, stream>>>(hbuf, ffnV, out_x, xcur, 1024, 1024, 4096);
}
